// SpectralTransform_1812476199046
// MI455X (gfx1250) — matrix operand order repaired and checked by reading, not compiled, not run on hardware
//
#include <hip/hip_runtime.h>
#include <hip/hip_bf16.h>
#include <cstdint>
#include <cstddef>

// ---------------------------------------------------------------------------
// FFC SpectralTransform on gfx1250 (MI455X), all heavy math as f16 WMMA GEMMs.
//   S0 conv1+bn+relu : (NHW x 256) @ (256 x 128)        -> y1   [n][c][h][w] f16
//   S1 FU fwd DFT    : rows(n,h,c) over w  @ Ff(128x160)-> ff   [n][h][t][2c+r]
//   S2 FU conv+bn+relu: rows(n,h,t) @ fu_wT(256x256)    -> z    [n][h][co][2t+r] (pad 160)
//   S3 FU inv DFT    : rows(n,h,co) @ G(160x128)        -> fuout[n][co][h][w]
//   S4 LFU fwd DFT   : rows(n,h',c3) over w' @ Ffl(64x96)-> ffl
//   S5 LFU conv      : rows(n,h',t) @ lfu_wT            -> zl (pad 96)
//   S6 LFU inv DFT   : rows(n,h',cl) @ Gl(96x64)        -> xsout[n][cl][h'][w']
//   S7 final conv    : w2(256x128) @ (y1+fuout+tile(xsout)) -> out NHWC f32
// B matrices zero-padded in N so every wave/WMMA is unconditional (EXEC full).
// A tile double-buffered in LDS (1 barrier / k-step); all WN B fragments are
// loaded before the WN back-to-back v_wmma ops so LOADcnt drains once.
// ---------------------------------------------------------------------------

typedef _Float16 v8h  __attribute__((ext_vector_type(8)));
typedef _Float16 v16h __attribute__((ext_vector_type(16)));
typedef float    v8f  __attribute__((ext_vector_type(8)));

#define ST_CONV1    0
#define ST_FU_FWD   1
#define ST_FU_CONV  2
#define ST_FU_INV   3
#define ST_LFU_FWD  4
#define ST_LFU_CONV 5
#define ST_LFU_INV  6
#define ST_FINAL    7

template<int S> constexpr int stK()   { return S==0?256: S==1?128: S==2?256: S==3?160: S==4?64: S==5?256: S==6?96: 128; }
template<int S> constexpr int stMW()  { return S==ST_FINAL ? 2 : 4; }              // waves in M
template<int S> constexpr int stWN()  { return S==0?4: S==1?5: S==2?8: S==3?4: S==4?3: S==5?8: S==6?2: 4; } // 16-col subtiles per wave
template<int S> constexpr int stLDB() { return S==0?128: S==1?160: S==2?256: S==3?128: S==4?96: S==5?256: S==6?64: 16; }

struct Args {
  const float* x;                 // input NHWC f32
  _Float16 *y1, *ff, *z, *fuout, *ffl, *zl, *xsout;
  const _Float16 *B;              // stage B matrix (K x LDB, row major, f16)
  const _Float16 *w2f;            // final-stage A (256 x 128 f16)
  const float *scale, *shift;     // folded BN for epilogue
  float* out;                     // final output NHWC f32
};

// ---- A-row base pointer (f16 sources), K-contiguous by construction --------
template<int S>
__device__ __forceinline__ const _Float16* a_row_ptr(const Args& a, int m) {
  if constexpr (S == ST_FU_FWD) {
    int c = m & 127, h = (m >> 7) & 127, n = m >> 14;
    return a.y1 + ((size_t)((n*128 + c)*128 + h)) * 128;
  } else if constexpr (S == ST_FU_CONV) {
    return a.ff + (size_t)m * 256;
  } else if constexpr (S == ST_FU_INV) {
    return a.z + (size_t)m * 160;
  } else if constexpr (S == ST_LFU_FWD) {
    int c3 = m & 127, hq = (m >> 7) & 63, n = m >> 13;
    int c = c3 & 31, sh = (c3 >> 5) & 1, sw = (c3 >> 6);
    return a.y1 + ((size_t)((n*128 + c)*128 + (sh*64 + hq))) * 128 + sw*64;
  } else if constexpr (S == ST_LFU_CONV) {
    return a.ffl + (size_t)m * 256;
  } else if constexpr (S == ST_LFU_INV) {
    return a.zl + (size_t)m * 96;
  } else { // ST_FINAL
    return a.w2f + (size_t)m * 128;
  }
}

// ---- cooperative A-tile staging (one k-step, MW*16 x 32 halves) ------------
template<int S, int MW>
__device__ __forceinline__ void stage_tile(const Args& a, int mbase, int k0,
                                           _Float16 (*buf)[32], int tid) {
  constexpr int PER = MW * 2;                 // halves per thread
  int e = tid * PER;
  int r = e >> 5, kk = e & 31;
  int m = mbase + r;
  if constexpr (S == ST_CONV1) {
    const float* p = a.x + (size_t)m * 256 + (k0 + kk);
#pragma unroll
    for (int q = 0; q < PER; ++q) buf[r][kk + q] = (_Float16)p[q];
  } else {
    const _Float16* p = a_row_ptr<S>(a, m) + (k0 + kk);
#pragma unroll
    for (int q = 0; q < PER; ++q) buf[r][kk + q] = p[q];
  }
}

// ---- A fragment from LDS per ISA 16-bit A 16x32 layout ---------------------
// lane L: row m = L&15; elems 0..7 -> K = ((L>>4)<<3)+0..7, elems 8..15 -> +16
__device__ __forceinline__ v16h frag_A(const _Float16 (*lds)[32], int lane) {
  int mm = lane & 15;
  int kh = (lane >> 4) << 3;
  union { v16h v; v8h h[2]; } u;
  u.h[0] = *(const v8h*)&lds[mm][kh];
  u.h[1] = *(const v8h*)&lds[mm][16 + kh];
  return u.v;
}

// ---- B fragment: lane L = col N=L&15, elems K per 16-bit table order --------
template<int S>
__device__ __forceinline__ v16h frag_B(const Args& a, int k0, int n0, int lane, int nb) {
  union { v16h v; v8h h[2]; _Float16 e[16]; } u;
  int kh = (lane >> 4) << 3;
  if constexpr (S != ST_FINAL) {
    const _Float16* p = a.B + (size_t)k0 * stLDB<S>() + n0 + (lane & 15);
#pragma unroll
    for (int i = 0; i < 16; ++i) {
      int k = (i < 8) ? (kh + i) : (16 + kh + (i - 8));
      u.e[i] = p[(size_t)k * stLDB<S>()];
    }
  } else {
    // B[c][hw] = y1 + fuout + tiled(xsout), built on the fly
    int h = n0 >> 7, w0 = (n0 & 127) + (lane & 15);
#pragma unroll
    for (int i = 0; i < 16; ++i) {
      int c = k0 + ((i < 8) ? (kh + i) : (16 + kh + (i - 8)));
      const _Float16* py = a.y1    + ((size_t)((nb*128 + c)*128 + h)) * 128 + w0;
      const _Float16* pf = a.fuout + ((size_t)((nb*128 + c)*128 + h)) * 128 + w0;
      const _Float16* px = a.xsout + ((size_t)((nb*128 + c)*64 + (h & 63))) * 64 + (w0 & 63);
      u.e[i] = (_Float16)((float)py[0] + (float)pf[0] + (float)px[0]);
    }
  }
  return u.v;
}

// ---- epilogue: per ISA 32-bit C/D layout: lane -> col = n0+(L&15),
//      VGPR g -> row = m0 + g + ((L>=16)?8:0) ---------------------------------
template<int S>
__device__ __forceinline__ void epilogue(const Args& a, int mtile, int ntile,
                                         int lane, const v8f& acc, int nb) {
  int col = ntile * 16 + (lane & 15);
  int rb  = mtile * 16 + ((lane >> 4) << 3);
#pragma unroll
  for (int g = 0; g < 8; ++g) {
    int m = rb + g;
    float v = acc[g];
    if constexpr (S == ST_CONV1) {
      v = v * a.scale[col] + a.shift[col];
      v = v > 0.f ? v : 0.f;
      int w = m & 127, h = (m >> 7) & 127, n = m >> 14;
      a.y1[((size_t)((n*128 + col)*128 + h)) * 128 + w] = (_Float16)v;
    } else if constexpr (S == ST_FU_FWD) {
      if (col < 130) {
        int c = m & 127, h = (m >> 7) & 127, n = m >> 14;
        a.ff[((size_t)((n*128 + h)*65 + (col >> 1))) * 256 + 2*c + (col & 1)] = (_Float16)v;
      }
    } else if constexpr (S == ST_FU_CONV) {
      v = v * a.scale[col] + a.shift[col];
      v = v > 0.f ? v : 0.f;
      int t = m % 65; int rem = m / 65; int h = rem & 127, n = rem >> 7;
      a.z[((size_t)((n*128 + h)*128 + (col >> 1))) * 160 + 2*t + (col & 1)] = (_Float16)v;
    } else if constexpr (S == ST_FU_INV) {
      int co = m & 127, h = (m >> 7) & 127, n = m >> 14;
      a.fuout[((size_t)((n*128 + co)*128 + h)) * 128 + col] = (_Float16)v;
    } else if constexpr (S == ST_LFU_FWD) {
      if (col < 66) {
        int c3 = m & 127, hq = (m >> 7) & 63, n = m >> 13;
        a.ffl[((size_t)((n*64 + hq)*33 + (col >> 1))) * 256 + 2*c3 + (col & 1)] = (_Float16)v;
      }
    } else if constexpr (S == ST_LFU_CONV) {
      v = v * a.scale[col] + a.shift[col];
      v = v > 0.f ? v : 0.f;
      int t = m % 33; int rem = m / 33; int hq = rem & 63, n = rem >> 6;
      a.zl[((size_t)((n*64 + hq)*128 + (col >> 1))) * 96 + 2*t + (col & 1)] = (_Float16)v;
    } else if constexpr (S == ST_LFU_INV) {
      int cl = m & 127, hq = (m >> 7) & 63, n = m >> 13;
      a.xsout[((size_t)((n*128 + cl)*64 + hq)) * 64 + col] = (_Float16)v;
    } else { // ST_FINAL : rows = out channel o, cols = hw; store NHWC f32
      a.out[((size_t)(nb*16384 + col)) * 256 + m] = v;
    }
  }
}

// ---- generic wave32 WMMA GEMM ----------------------------------------------
// Block = 8 waves arranged MW x NW; each wave owns WN 16-col subtiles.
// Double-buffered LDS A tile: stage k+1 while computing k, 1 barrier/k-step.
// All WN B fragments loaded before the WMMA burst.
template<int S>
__global__ __launch_bounds__(256) void gemm_k(Args a) {
  constexpr int MW = stMW<S>();
  constexpr int NW = 8 / MW;
  constexpr int WN = stWN<S>();
  constexpr int K  = stK<S>();
  constexpr int NK = K / 32;
  __shared__ alignas(32) _Float16 lds[2][MW * 16][32];

  const int tid  = threadIdx.x;
  const int lane = tid & 31;
  const int wid  = tid >> 5;
  const int mw   = wid & (MW - 1);
  const int nw   = wid / MW;
  const int mbase = blockIdx.x * (MW * 16);   // first row of this block
  const int nbase = blockIdx.y * (NW * WN * 16);
  const int nb    = blockIdx.z;
  const int nw0   = nbase + nw * WN * 16;     // first col of this wave

  v8f acc[WN];
#pragma unroll
  for (int j = 0; j < WN; ++j)
#pragma unroll
    for (int i = 0; i < 8; ++i) acc[j][i] = 0.f;

  stage_tile<S, MW>(a, mbase, 0, lds[0], tid);
  __syncthreads();

  for (int ks = 0; ks < NK; ++ks) {
    const int cur = ks & 1;
    if (ks + 1 < NK)                          // stage next tile into ping buffer
      stage_tile<S, MW>(a, mbase, (ks + 1) * 32, lds[cur ^ 1], tid);
    if constexpr (S != ST_FINAL) {
      if (ks + 2 < NK)
        __builtin_prefetch(a.B + (size_t)((ks + 2) * 32 + lane) * stLDB<S>() + nw0, 0, 3);
    }
    v16h fa = frag_A((const _Float16 (*)[32])&lds[cur][mw * 16], lane);
    v16h fb[WN];
#pragma unroll
    for (int j = 0; j < WN; ++j)
      fb[j] = frag_B<S>(a, ks * 32, nw0 + j * 16, lane, nb);
#pragma unroll
    for (int j = 0; j < WN; ++j)
      acc[j] = __builtin_amdgcn_wmma_f32_16x16x32_f16(false, fa, false, fb[j],
                                                      (short)0, acc[j], false, false);
    __syncthreads();                          // fences ping write + pong reuse
  }
#pragma unroll
  for (int j = 0; j < WN; ++j)
    epilogue<S>(a, blockIdx.x * MW + mw, (nw0 >> 4) + j, lane, acc[j], nb);
}

// ---------------------------- prep kernels ----------------------------------
__global__ void k_prep_wT(const float* w1, const float* fuw, const float* lfuw,
                          const float* w2, _Float16* w1T, _Float16* fuwT,
                          _Float16* lfuwT, _Float16* w2f) {
  int i = blockIdx.x * 256 + threadIdx.x;          // 65536 threads
  if (i < 32768) {
    int c = i >> 7, o = i & 127;
    w1T[i] = (_Float16)w1[o*256 + c];              // w1T[c][o]
    w2f[i] = (_Float16)w2[i];                      // w2f[o][c] (row-major copy)
  }
  int j = i >> 8, o = i & 255;                     // i < 65536 always
  fuwT[i]  = (_Float16)fuw[o*256 + j];             // fu_wT[j][o]
  lfuwT[i] = (_Float16)lfuw[o*256 + j];
}

__global__ void k_prep_dft(_Float16* Ff, _Float16* G, _Float16* Ffl, _Float16* Gl) {
  int i = blockIdx.x * 256 + threadIdx.x;          // >= 20480 threads
  const float TP = 6.28318530718f;
  if (i < 128*160) {                               // Ff[w][j], j=2t+r, /sqrt(128)
    int w = i / 160, j = i % 160;
    float v = 0.f;
    if (j < 130) {
      int t = j >> 1;
      float ang = TP * (float)((w*t) & 127) * (1.f/128.f);
      v = ((j & 1) ? -__sinf(ang) : __cosf(ang)) * 0.08838834765f;
    }
    Ff[i] = (_Float16)v;
  }
  if (i < 160*128) {                               // G[j][w], wt*cos/-sin /sqrt(128)
    int j = i / 128, w = i % 128;
    float v = 0.f;
    if (j < 130) {
      int t = j >> 1;
      float wt = (t == 0 || t == 64) ? 1.f : 2.f;
      float ang = TP * (float)((w*t) & 127) * (1.f/128.f);
      v = ((j & 1) ? -__sinf(ang) : __cosf(ang)) * wt * 0.08838834765f;
    }
    G[i] = (_Float16)v;
  }
  if (i < 64*96) {                                 // Ffl[w][j], /8
    int w = i / 96, j = i % 96;
    float v = 0.f;
    if (j < 66) {
      int t = j >> 1;
      float ang = TP * (float)((w*t) & 63) * (1.f/64.f);
      v = ((j & 1) ? -__sinf(ang) : __cosf(ang)) * 0.125f;
    }
    Ffl[i] = (_Float16)v;
  }
  if (i < 96*64) {                                 // Gl[j][w], wt/8
    int j = i / 64, w = i % 64;
    float v = 0.f;
    if (j < 66) {
      int t = j >> 1;
      float wt = (t == 0 || t == 32) ? 1.f : 2.f;
      float ang = TP * (float)((w*t) & 63) * (1.f/64.f);
      v = ((j & 1) ? -__sinf(ang) : __cosf(ang)) * wt * 0.125f;
    }
    Gl[i] = (_Float16)v;
  }
}

__global__ void k_prep_bn(const float* g1, const float* b1, const float* m1, const float* v1,
                          const float* fg, const float* fb, const float* fm, const float* fv,
                          const float* lg, const float* lb, const float* lm, const float* lv,
                          float* s1, float* sh1, float* fs, float* fsh, float* ls, float* lsh) {
  int i = threadIdx.x;                             // 256 threads, 1 block
  if (i < 128) {
    float s = g1[i] * rsqrtf(v1[i] + 1e-3f);
    s1[i] = s; sh1[i] = b1[i] - m1[i] * s;
  }
  float s = fg[i] * rsqrtf(fv[i] + 1e-3f);
  fs[i] = s; fsh[i] = fb[i] - fm[i] * s;
  float t = lg[i] * rsqrtf(lv[i] + 1e-3f);
  ls[i] = t; lsh[i] = lb[i] - lm[i] * t;
}

__global__ void k_zero_h(_Float16* p, long n) {
  long i = (long)blockIdx.x * blockDim.x + threadIdx.x;
  long stride = (long)gridDim.x * blockDim.x;
  for (; i < n; i += stride) p[i] = (_Float16)0.f;
}

// ----------------------------------------------------------------------------
extern "C" void kernel_launch(void* const* d_in, const int* in_sizes, int n_in,
                              void* d_out, int out_size, void* d_ws, size_t ws_size,
                              hipStream_t stream) {
  const float* x    = (const float*)d_in[0];
  const float* w1   = (const float*)d_in[1];
  const float* g1   = (const float*)d_in[2];
  const float* b1   = (const float*)d_in[3];
  const float* m1   = (const float*)d_in[4];
  const float* v1   = (const float*)d_in[5];
  const float* fuw  = (const float*)d_in[6];
  const float* fug  = (const float*)d_in[7];
  const float* fub  = (const float*)d_in[8];
  const float* fum  = (const float*)d_in[9];
  const float* fuv  = (const float*)d_in[10];
  const float* lfw  = (const float*)d_in[11];
  const float* lfg  = (const float*)d_in[12];
  const float* lfb  = (const float*)d_in[13];
  const float* lfm  = (const float*)d_in[14];
  const float* lfv  = (const float*)d_in[15];
  const float* w2   = (const float*)d_in[16];
  float* out = (float*)d_out;

  size_t off = 0;
  auto a16 = [&](size_t elems) -> _Float16* {
    off = (off + 255) & ~(size_t)255;
    _Float16* p = (_Float16*)((char*)d_ws + off);
    off += elems * sizeof(_Float16);
    return p;
  };
  auto a32 = [&](size_t elems) -> float* {
    off = (off + 255) & ~(size_t)255;
    float* p = (float*)((char*)d_ws + off);
    off += elems * sizeof(float);
    return p;
  };

  _Float16* y1    = a16((size_t)8*128*128*128);    // 33.5 MB
  _Float16* ff    = a16((size_t)8*128*65*256);     // 34.1 MB
  _Float16* z     = a16((size_t)8*128*128*160);    // 41.9 MB
  _Float16* fuout = a16((size_t)8*128*128*128);    // 33.5 MB
  _Float16* ffl   = a16((size_t)8*64*33*256);      //  8.7 MB
  _Float16* zl    = a16((size_t)8*64*128*96);      // 12.6 MB
  _Float16* xsout = a16((size_t)8*128*64*64);      //  8.4 MB
  _Float16* w1T   = a16(256*128);
  _Float16* Ff    = a16(128*160);
  _Float16* fuwT  = a16(256*256);
  _Float16* G     = a16(160*128);
  _Float16* Ffl   = a16(64*96);
  _Float16* lfuwT = a16(256*256);
  _Float16* Gl    = a16(96*64);
  _Float16* w2f   = a16(256*128);
  float* s1  = a32(128); float* sh1 = a32(128);
  float* fs  = a32(256); float* fsh = a32(256);
  float* ls  = a32(256); float* lsh = a32(256);

  // --- constant-table prep (deterministic every call) ---
  k_prep_wT <<<256, 256, 0, stream>>>(w1, fuw, lfw, w2, w1T, fuwT, lfuwT, w2f);
  k_prep_dft<<<80,  256, 0, stream>>>(Ff, G, Ffl, Gl);
  k_prep_bn <<<1,   256, 0, stream>>>(g1, b1, m1, v1, fug, fub, fum, fuv,
                                      lfg, lfb, lfm, lfv, s1, sh1, fs, fsh, ls, lsh);
  k_zero_h  <<<2048, 256, 0, stream>>>(z,  (long)8*128*128*160);
  k_zero_h  <<<1024, 256, 0, stream>>>(zl, (long)8*64*128*96);

  Args a{};
  a.x = x; a.y1 = y1; a.ff = ff; a.z = z; a.fuout = fuout;
  a.ffl = ffl; a.zl = zl; a.xsout = xsout; a.w2f = w2f; a.out = out;

  // S0: conv1 + bn + relu                   M=131072 (8192 tiles / MW=4)
  a.B = w1T; a.scale = s1; a.shift = sh1;
  gemm_k<ST_CONV1><<<dim3(2048, 1, 1), 256, 0, stream>>>(a);
  // S1: FU forward DFT                      M=131072
  a.B = Ff;
  gemm_k<ST_FU_FWD><<<dim3(2048, 1, 1), 256, 0, stream>>>(a);
  // S2: FU 1x1 conv + bn + relu             M=66560 (4160 tiles / 4)
  a.B = fuwT; a.scale = fs; a.shift = fsh;
  gemm_k<ST_FU_CONV><<<dim3(1040, 1, 1), 256, 0, stream>>>(a);
  // S3: FU inverse DFT                      M=131072
  a.B = G;
  gemm_k<ST_FU_INV><<<dim3(2048, 1, 1), 256, 0, stream>>>(a);
  // S4: LFU forward DFT                     M=65536 (4096 tiles / 4)
  a.B = Ffl;
  gemm_k<ST_LFU_FWD><<<dim3(1024, 1, 1), 256, 0, stream>>>(a);
  // S5: LFU conv + bn + relu                M=16896 (1056 tiles / 4)
  a.B = lfuwT; a.scale = ls; a.shift = lsh;
  gemm_k<ST_LFU_CONV><<<dim3(264, 1, 1), 256, 0, stream>>>(a);
  // S6: LFU inverse DFT                     M=65536
  a.B = Gl;
  gemm_k<ST_LFU_INV><<<dim3(1024, 1, 1), 256, 0, stream>>>(a);
  // S7: final conv -> NHWC f32              M=256 (16 tiles / MW=2), N=16384
  gemm_k<ST_FINAL><<<dim3(8, 64, 8), 256, 0, stream>>>(a);

  (void)in_sizes; (void)n_in; (void)out_size; (void)ws_size;
}
